// LeakyIntegrationAttention_33157147525444
// MI455X (gfx1250) — matrix operand order repaired and checked by reading, not compiled, not run on hardware
//
#include <hip/hip_runtime.h>

// ---------------- problem constants ----------------
constexpr int BB  = 4;
constexpr int SS  = 2048;
constexpr int DD  = 512;
constexpr int HH  = 8;
constexpr int DHH = 64;               // head dim
constexpr float SCALE = 0.04419417382415922f; // 1/sqrt(512)

// ---------------- WMMA / TDM types ----------------
typedef __attribute__((ext_vector_type(16))) __bf16 v16bf;
typedef __attribute__((ext_vector_type(8)))  float  v8f;
typedef __attribute__((ext_vector_type(4)))  unsigned int v4u32;
typedef __attribute__((ext_vector_type(8)))  int v8i32;
typedef __attribute__((ext_vector_type(4)))  int v4i32;

union FragBF { v16bf v; unsigned short u[16]; };

#if defined(__has_builtin)
#if __has_builtin(__builtin_amdgcn_tensor_load_to_lds)
#define HAVE_TDM 1
#endif
#endif

#ifdef HAVE_TDM
#define TDM_WAIT(n) __builtin_amdgcn_s_wait_tensorcnt(n)
#else
#define TDM_WAIT(n)
#endif

// 2-D tile DMA: nrows rows of row_dw dwords, global row pitch stride_dw dwords,
// packed contiguously at LDS byte offset lds_off. Wave-uniform args.
__device__ __forceinline__ void tdm_load_2d(const void* gsrc, unsigned lds_off,
                                            unsigned row_dw, unsigned nrows,
                                            unsigned stride_dw) {
#ifdef HAVE_TDM
  unsigned long long ga = (unsigned long long)(size_t)gsrc;
  v4u32 g0;
  g0[0] = 1u;                                   // count=1, user descriptor
  g0[1] = lds_off;                              // LDS byte address
  g0[2] = (unsigned)ga;                         // global_addr[31:0]
  g0[3] = (unsigned)((ga >> 32) & 0x1FFFFFFu) | (2u << 30);  // addr[56:32] | type=2
  v8i32 g1;
  g1[0] = (int)(2u << 16);                      // workgroup_mask=0, data_size=4B
  g1[1] = (int)((row_dw & 0xFFFFu) << 16);      // tensor_dim0[15:0]
  g1[2] = (int)((row_dw >> 16) | ((nrows & 0xFFFFu) << 16));  // dim0[31:16] | dim1[15:0]
  g1[3] = (int)((nrows >> 16) | (row_dw << 16));// dim1[31:16] | tile_dim0
  g1[4] = (int)nrows;                           // tile_dim1 (tile_dim2=0)
  g1[5] = (int)stride_dw;                       // tensor_dim0_stride[31:0]
  g1[6] = 0;                                    // stride[47:32]=0, dim1_stride lo=0
  g1[7] = 0;
  v4i32 g2; g2[0] = 1; g2[1] = 1; g2[2] = 0; g2[3] = 0;  // dim2=1, dim3=1, tile_dim3=0
  v4i32 g3; g3[0] = 0; g3[1] = 1; g3[2] = 0; g3[3] = 0;  // dim3_stride=0, dim4=1, tile_dim4=0
#if __clang_major__ >= 23
  v8i32 g4; g4[0]=0; g4[1]=0; g4[2]=0; g4[3]=0; g4[4]=0; g4[5]=0; g4[6]=0; g4[7]=0;
  __builtin_amdgcn_tensor_load_to_lds(g0, g1, g2, g3, g4, 0);
#else
  __builtin_amdgcn_tensor_load_to_lds(g0, g1, g2, g3, 0);
#endif
#else
  (void)gsrc; (void)lds_off; (void)row_dw; (void)nrows; (void)stride_dw;
#endif
}

// stage one 64x64 bf16 tile (64 rows x 128B): each wave DMAs its 8-row slice
__device__ __forceinline__ void stage_tile64(unsigned short* dst,
                                             const unsigned short* gsrc,
                                             int wv, int tid) {
#ifdef HAVE_TDM
  tdm_load_2d(gsrc + (size_t)wv * 8 * DHH,
              (unsigned)(size_t)dst + (unsigned)(wv * 8 * DHH * 2),
              /*row_dw=*/DHH / 2, /*nrows=*/8, /*stride_dw=*/DHH / 2);
#else
  const uint4* s = (const uint4*)gsrc;
  uint4*       d = (uint4*)dst;
  for (int e = tid; e < 512; e += 256) d[e] = s[e];
  (void)wv;
#endif
}

__device__ __forceinline__ unsigned short f2bf(float f) {
  unsigned int u = __float_as_uint(f);
  u += 0x7FFFu + ((u >> 16) & 1u);    // round-to-nearest-even
  return (unsigned short)(u >> 16);
}

__device__ __forceinline__ float redmax16(float x) {
  x = fmaxf(x, __shfl_xor(x, 1, 32));
  x = fmaxf(x, __shfl_xor(x, 2, 32));
  x = fmaxf(x, __shfl_xor(x, 4, 32));
  x = fmaxf(x, __shfl_xor(x, 8, 32));
  return x;
}
__device__ __forceinline__ float redsum16(float x) {
  x += __shfl_xor(x, 1, 32);
  x += __shfl_xor(x, 2, 32);
  x += __shfl_xor(x, 4, 32);
  x += __shfl_xor(x, 8, 32);
  return x;
}

// A-frag K index (16-bit A 16x32 layout), B-frag K index (16-bit B 32x16 layout)
__device__ __forceinline__ int kA(int i, int hi) { return i + ((i >= 8) ? 8 : 0) + 8 * hi; }
__device__ __forceinline__ int kB(int i, int hi) { return i + ((i >= 8) ? 8 : 0) + 8 * hi; }

// =====================================================================
// GEMM: Y[m, j] = sum_k X[m,k] * W[j,k] + bias[j]   (X: 8192x512, W: 512x512)
// out_mode 0: f32, flat (B,S,D)     out_mode 1: bf16, head layout (B,H,S,dh)
// =====================================================================
__global__ __launch_bounds__(256)
void gemm_bias_kernel(const void* __restrict__ Xv, const float* __restrict__ W,
                      const float* __restrict__ bias, void* __restrict__ Yv,
                      int in_bf16, int negate, int out_mode)
{
  __shared__ unsigned short Xl[128 * 32];  // 8 KB
  __shared__ unsigned short Wl[64 * 32];   // 4 KB

  const int tid  = threadIdx.x;
  const int wv   = tid >> 5;
  const int lane = tid & 31;
  const int hi   = (lane >> 4) & 1;
  const int ln   = lane & 15;
  const int m0   = blockIdx.x * 128;
  const int n0   = blockIdx.y * 64;

  v8f acc[4] = {};
  const float*          Xf = (const float*)Xv;
  const unsigned short* Xb = (const unsigned short*)Xv;

  for (int kc = 0; kc < DD / 32; ++kc) {
    if (in_bf16) {
#ifdef HAVE_TDM
      // wave wv DMAs its 16-row slice (16 rows x 64B), row pitch 1024B
      tdm_load_2d(Xb + (size_t)(m0 + wv * 16) * DD + kc * 32,
                  (unsigned)(size_t)Xl + (unsigned)(wv * 16 * 64),
                  /*row_dw=*/16, /*nrows=*/16, /*stride_dw=*/DD / 2);
#else
      for (int e = tid; e < 512; e += 256) {      // 128 rows * 4 uint4
        int r = e >> 2, c4 = e & 3;
        const uint4* srow = (const uint4*)(Xb + (size_t)(m0 + r) * DD + kc * 32);
        ((uint4*)Xl)[e] = srow[c4];
      }
#endif
    } else {
      for (int e = tid; e < 2048; e += 256) {     // paired f32 -> packed bf16x2
        int r = e >> 4, c2 = e & 15;
        float2 xv = ((const float2*)(Xf + (size_t)(m0 + r) * DD + kc * 32))[c2];
        ((unsigned int*)Xl)[e] =
            (unsigned)f2bf(xv.x) | ((unsigned)f2bf(xv.y) << 16);
      }
    }
    for (int e = tid; e < 1024; e += 256) {
      int r = e >> 4, c2 = e & 15;
      float2 wv2 = ((const float2*)(W + (size_t)(n0 + r) * DD + kc * 32))[c2];
      ((unsigned int*)Wl)[e] =
          (unsigned)f2bf(wv2.x) | ((unsigned)f2bf(wv2.y) << 16);
    }
    if (in_bf16) { TDM_WAIT(0); }
    __syncthreads();

    FragBF a;
#pragma unroll
    for (int i = 0; i < 16; ++i) a.u[i] = Xl[(wv * 16 + ln) * 32 + kA(i, hi)];

#pragma unroll
    for (int t = 0; t < 4; ++t) {
      FragBF bf;
#pragma unroll
      for (int i = 0; i < 16; ++i) bf.u[i] = Wl[(t * 16 + ln) * 32 + kB(i, hi)];
      acc[t] = __builtin_amdgcn_wmma_f32_16x16x32_bf16(
          false, a.v, false, bf.v, (short)0, acc[t], false, false);
    }
    __syncthreads();
  }

#pragma unroll
  for (int t = 0; t < 4; ++t) {
    int   j  = n0 + t * 16 + ln;
    float bj = bias[j];
#pragma unroll
    for (int r = 0; r < 8; ++r) {
      int   m   = m0 + wv * 16 + r + 8 * hi;
      float val = acc[t][r] + bj;
      if (negate) val = -val;                       // leaky integrate: x -> -x
      if (out_mode == 0) {
        ((float*)Yv)[(size_t)m * DD + j] = val;
      } else {
        int b = m / SS, s = m % SS;
        int h = j / DHH, d = j % DHH;
        ((unsigned short*)Yv)[(((size_t)b * HH + h) * SS + s) * DHH + d] = f2bf(val);
      }
    }
  }
}

// =====================================================================
// Attention: per block = one (b,h, 128-row q tile); two-pass softmax+sigmoid
// q,k,v: bf16 (B,H,S,dh); o: bf16 (B,S,D). TDM double-buffered K/V staging.
// =====================================================================
__global__ __launch_bounds__(256)
void attn_kernel(const unsigned short* __restrict__ q,
                 const unsigned short* __restrict__ k,
                 const unsigned short* __restrict__ v,
                 unsigned short* __restrict__ o)
{
  __shared__ unsigned short Kl[2][64 * DHH];     // 2 x 8 KB
  __shared__ unsigned short Vl[2][64 * DHH];     // 2 x 8 KB
  __shared__ unsigned short Astage[8 * 16 * 32]; // 8 KB, per-wave C->A relayout

  const int tid  = threadIdx.x;
  const int wv   = tid >> 5;
  const int lane = tid & 31;
  const int hi   = (lane >> 4) & 1;
  const int ln   = lane & 15;
  const int bh   = blockIdx.x >> 4;   // S/128 = 16 q tiles
  const int qt   = blockIdx.x & 15;
  const int q0   = qt * 128 + wv * 16;
  constexpr int NC = SS / 64;

  // Q fragments for this wave's 16 rows: two 16x32 A-frags over dh=64
  FragBF qf[2];
  {
    const unsigned short* qrow = q + ((size_t)bh * SS + q0 + ln) * DHH;
#pragma unroll
    for (int f = 0; f < 2; ++f)
#pragma unroll
      for (int i = 0; i < 16; ++i) qf[f].u[i] = qrow[kA(i, hi) + 32 * f];
  }

  float m[8], Z[8];
#pragma unroll
  for (int r = 0; r < 8; ++r) { m[r] = -3.0e38f; Z[r] = 0.0f; }

  const size_t kvbase = (size_t)bh * SS * DHH;

  // ---------------- Pass A: exact row max + sum of exp ----------------
  int buf = 0;
  stage_tile64(Kl[0], k + kvbase, wv, tid);
  for (int ck = 0; ck < NC; ++ck) {
    if (ck + 1 < NC) {
      stage_tile64(Kl[buf ^ 1], k + kvbase + (size_t)(ck + 1) * 64 * DHH, wv, tid);
      TDM_WAIT(1);                 // chunk ck's DMA retired (in-order per wave)
    } else {
      TDM_WAIT(0);
    }
    __syncthreads();

    const unsigned short* KB = Kl[buf];
    float sv[4][8];
#pragma unroll
    for (int t = 0; t < 4; ++t) {
      FragBF kb0, kb1;
#pragma unroll
      for (int i = 0; i < 16; ++i) {
        kb0.u[i] = KB[(t * 16 + ln) * DHH + kB(i, hi)];
        kb1.u[i] = KB[(t * 16 + ln) * DHH + 32 + kB(i, hi)];
      }
      v8f c = {};
      c = __builtin_amdgcn_wmma_f32_16x16x32_bf16(false, qf[0].v, false, kb0.v, (short)0, c, false, false);
      c = __builtin_amdgcn_wmma_f32_16x16x32_bf16(false, qf[1].v, false, kb1.v, (short)0, c, false, false);
#pragma unroll
      for (int r = 0; r < 8; ++r) sv[t][r] = c[r] * SCALE;
    }
#pragma unroll
    for (int r = 0; r < 8; ++r) {
      float cm = fmaxf(fmaxf(sv[0][r], sv[1][r]), fmaxf(sv[2][r], sv[3][r]));
      cm = redmax16(cm);
      float mn = fmaxf(m[r], cm);
      float zs = __expf(m[r] - mn);
      float ad = __expf(sv[0][r] - mn) + __expf(sv[1][r] - mn) +
                 __expf(sv[2][r] - mn) + __expf(sv[3][r] - mn);
      Z[r] = Z[r] * zs + ad;   // lane-partial; reduced after the loop
      m[r] = mn;               // already lane-uniform within row group
    }
    __syncthreads();           // all waves done with Kl[buf] before restage
    buf ^= 1;
  }

  float invZ[8];
#pragma unroll
  for (int r = 0; r < 8; ++r) invZ[r] = 1.0f / redsum16(Z[r]);

  v8f oacc[4] = {};

  // ---------------- Pass B: recompute, sigmoid(softmax), O += A*V ----------------
  buf = 0;
  stage_tile64(Kl[0], k + kvbase, wv, tid);
  stage_tile64(Vl[0], v + kvbase, wv, tid);
  for (int ck = 0; ck < NC; ++ck) {
    if (ck + 1 < NC) {
      stage_tile64(Kl[buf ^ 1], k + kvbase + (size_t)(ck + 1) * 64 * DHH, wv, tid);
      stage_tile64(Vl[buf ^ 1], v + kvbase + (size_t)(ck + 1) * 64 * DHH, wv, tid);
      TDM_WAIT(2);               // both DMAs of chunk ck retired
    } else {
      TDM_WAIT(0);
    }
    __syncthreads();

    const unsigned short* KB = Kl[buf];
    const unsigned short* VB = Vl[buf];
#pragma unroll
    for (int p = 0; p < 2; ++p) {                 // two 16x32 A-tiles per chunk
#pragma unroll
      for (int tt = 0; tt < 2; ++tt) {
        int t = p * 2 + tt;
        FragBF kb0, kb1;
#pragma unroll
        for (int i = 0; i < 16; ++i) {
          kb0.u[i] = KB[(t * 16 + ln) * DHH + kB(i, hi)];
          kb1.u[i] = KB[(t * 16 + ln) * DHH + 32 + kB(i, hi)];
        }
        v8f c = {};
        c = __builtin_amdgcn_wmma_f32_16x16x32_bf16(false, qf[0].v, false, kb0.v, (short)0, c, false, false);
        c = __builtin_amdgcn_wmma_f32_16x16x32_bf16(false, qf[1].v, false, kb1.v, (short)0, c, false, false);
#pragma unroll
        for (int r = 0; r < 8; ++r) {
          float s  = c[r] * SCALE;
          float pv = __expf(s - m[r]) * invZ[r];        // softmax prob
          float av = 1.0f / (1.0f + __expf(-pv));       // sigmoid after softmax
          Astage[wv * 512 + (r + 8 * hi) * 32 + tt * 16 + ln] = f2bf(av);
        }
      }
      // Astage is per-wave private; DS ops are in-order within a wave -> no barrier
      FragBF af;
#pragma unroll
      for (int i = 0; i < 16; ++i) af.u[i] = Astage[wv * 512 + ln * 32 + kA(i, hi)];

#pragma unroll
      for (int nd = 0; nd < 4; ++nd) {            // dh = 4 N-tiles of 16
        FragBF vb;
#pragma unroll
        for (int i = 0; i < 16; ++i)
          vb.u[i] = VB[(p * 32 + kB(i, hi)) * DHH + nd * 16 + ln];
        oacc[nd] = __builtin_amdgcn_wmma_f32_16x16x32_bf16(
            false, af.v, false, vb.v, (short)0, oacc[nd], false, false);
      }
    }
    __syncthreads();           // all waves done with Kl/Vl[buf] before restage
    buf ^= 1;
  }

  // write O as bf16 in (B,S,D), merging heads
  const int b = bh / HH, h = bh % HH;
#pragma unroll
  for (int nd = 0; nd < 4; ++nd)
#pragma unroll
    for (int r = 0; r < 8; ++r) {
      int srow = q0 + r + 8 * hi;
      int d    = h * DHH + nd * 16 + ln;
      o[((size_t)b * SS + srow) * DD + d] = f2bf(oacc[nd][r]);
    }
}

// =====================================================================
extern "C" void kernel_launch(void* const* d_in, const int* in_sizes, int n_in,
                              void* d_out, int out_size, void* d_ws, size_t ws_size,
                              hipStream_t stream) {
  (void)in_sizes; (void)n_in; (void)out_size; (void)ws_size;
  const float* query = (const float*)d_in[0];
  const float* key_  = (const float*)d_in[1];
  const float* value = (const float*)d_in[2];
  const float* Wq = (const float*)d_in[3];
  const float* bq = (const float*)d_in[4];
  const float* Wk = (const float*)d_in[5];
  const float* bk = (const float*)d_in[6];
  const float* Wv = (const float*)d_in[7];
  const float* bv = (const float*)d_in[8];
  const float* Wo = (const float*)d_in[9];
  const float* bo = (const float*)d_in[10];
  float* out = (float*)d_out;

  char*  ws  = (char*)d_ws;
  size_t per = (size_t)BB * HH * SS * DHH * sizeof(unsigned short); // 8 MiB each
  unsigned short* qws = (unsigned short*)(ws);
  unsigned short* kws = (unsigned short*)(ws + per);
  unsigned short* vws = (unsigned short*)(ws + 2 * per);
  unsigned short* ows = (unsigned short*)(ws + 3 * per);

  dim3 gg(BB * SS / 128, DD / 64);  // 64 x 8 blocks
  gemm_bias_kernel<<<gg, 256, 0, stream>>>(query, Wq, bq, qws, 0, 1, 1);
  gemm_bias_kernel<<<gg, 256, 0, stream>>>(key_,  Wk, bk, kws, 0, 1, 1);
  gemm_bias_kernel<<<gg, 256, 0, stream>>>(value, Wv, bv, vws, 0, 0, 1);
  attn_kernel<<<BB * HH * (SS / 128), 256, 0, stream>>>(qws, kws, vws, ows);
  gemm_bias_kernel<<<gg, 256, 0, stream>>>(ows, Wo, bo, out, 1, 0, 0);
}
